// HTNetWithNCDE_62783831933531
// MI455X (gfx1250) — matrix operand order repaired and checked by reading, not compiled, not run on hardware
//
#include <hip/hip_runtime.h>
#include <math.h>

typedef __attribute__((ext_vector_type(16))) _Float16 v16h;
typedef __attribute__((ext_vector_type(8)))  float    v8f;

#define BN_EPS 1e-5f

__device__ __forceinline__ v8f wmma16(v16h a, v16h b, v8f c) {
  // D = A(16x32 f16) * B(32x16 f16) + C(16x16 f32)
  return __builtin_amdgcn_wmma_f32_16x16x32_f16(false, a, false, b, (short)0, c, false, false);
}

__device__ __forceinline__ float fast_exp(float x) {
#if __has_builtin(__builtin_amdgcn_exp2f)
  return __builtin_amdgcn_exp2f(x * 1.4426950408889634f);
#else
  return exp2f(x * 1.4426950408889634f);
#endif
}

__device__ __forceinline__ float fast_rcp(float x) {
#if __has_builtin(__builtin_amdgcn_rcpf)
  return __builtin_amdgcn_rcpf(x);
#else
  return 1.f / x;
#endif
}

// Branch-free tanh: prefer HW V_TANH_F32 (CDNA5 TRANS op), else exp2+rcp.
__device__ __forceinline__ float fast_tanh(float x) {
#if __has_builtin(__builtin_amdgcn_tanhf)
  return __builtin_amdgcn_tanhf(x);
#elif __has_builtin(__builtin_amdgcn_tanh_f32)
  return __builtin_amdgcn_tanh_f32(x);
#else
  float ax = fabsf(x);
  float u = fast_exp(-2.f * ax);            // e^(-2|x|)
  float t = (1.f - u) * fast_rcp(1.f + u);  // tanh(|x|)
  return copysignf(t, x);
#endif
}

// Branch-free ELU (alpha=1): x>0 ? x : e^x - 1
__device__ __forceinline__ float eluf(float x) {
  float e = fast_exp(fminf(x, 0.f)) - 1.f;
  return x > 0.f ? x : e;
}

// ---------------------------------------------------------------------------
// Small prep: weight-normalized matrices + fused conv2 projection
// V1n = d1_g * d1_v / ||row||           (64 x 32)
// wr  = dw_g * dw_v / ||row||           (16 x 64)
// V2n = d2_g * d2_v / ||row||           (16 x 32)
// M2  = conv2_w @ V2n                   (16 x 32)
// e2  = conv2_w @ d2_b + conv2_b        (16)
// ---------------------------------------------------------------------------
__global__ __launch_bounds__(256) void k_prep_small(
    const float* __restrict__ d1_g, const float* __restrict__ d1_v,
    const float* __restrict__ dw_g, const float* __restrict__ dw_v,
    const float* __restrict__ d2_g, const float* __restrict__ d2_v,
    const float* __restrict__ conv2_w, const float* __restrict__ d2_b,
    const float* __restrict__ conv2_b,
    float* __restrict__ V1n, float* __restrict__ wr,
    float* __restrict__ V2n, float* __restrict__ M2, float* __restrict__ e2)
{
  int t = threadIdx.x;
  if (t < 64) {
    float ss = 0.f;
    for (int h = 0; h < 32; ++h) { float v = d1_v[t*32+h]; ss += v*v; }
    float sc = d1_g[t] / sqrtf(ss);
    for (int h = 0; h < 32; ++h) V1n[t*32+h] = d1_v[t*32+h] * sc;
  }
  if (t < 16) {
    float ss = 0.f;
    for (int c = 0; c < 64; ++c) { float v = dw_v[t*64+c]; ss += v*v; }
    float sc = dw_g[t] / sqrtf(ss);
    for (int c = 0; c < 64; ++c) wr[t*64+c] = dw_v[t*64+c] * sc;
  }
  if (t < 16) {
    float ss = 0.f;
    for (int h = 0; h < 32; ++h) { float v = d2_v[t*32+h]; ss += v*v; }
    float sc = d2_g[t] / sqrtf(ss);
    for (int h = 0; h < 32; ++h) V2n[t*32+h] = d2_v[t*32+h] * sc;
  }
  __syncthreads();
  for (int i = t; i < 512; i += 256) {
    int o = i >> 5, h = i & 31;
    float acc = 0.f;
    for (int c = 0; c < 16; ++c) acc += conv2_w[o*16+c] * V2n[c*32+h];
    M2[i] = acc;
  }
  if (t < 16) {
    float acc = 0.f;
    for (int c = 0; c < 16; ++c) acc += conv2_w[t*16+c] * d2_b[c];
    e2[t] = acc + conv2_b[t];
  }
}

// ---------------------------------------------------------------------------
// Swizzle a row-major (N x K) f32 matrix into WMMA 16x16x32 B-fragment tiles:
// dst[(kt*NT+nt)*512 + lane*16 + j] = f16( src[(nt*16 + lane%16)*K + kt*32 + j + 16*(lane/16)] )
// ---------------------------------------------------------------------------
__global__ __launch_bounds__(256) void k_swizzleB(
    const float* __restrict__ src, _Float16* __restrict__ dst, int N, int K)
{
  int NT = N >> 4;
  int KT = K >> 5;
  int total = NT * KT * 512;
  int idx = blockIdx.x * 256 + threadIdx.x;
  if (idx >= total) return;
  int tile = idx >> 9;
  int l = (idx >> 4) & 31;
  int j = idx & 15;
  int kt = tile / NT, nt = tile % NT;
  int n = nt * 16 + (l & 15);
  int k = kt * 32 + j + ((l >> 4) << 4);
  dst[idx] = (_Float16)src[(size_t)n * K + k];
}

// ---------------------------------------------------------------------------
// Build time-augmented, time-major input for NCDE1: xa1[s][b][c], c==0 -> s*dt
// x layout: (B=64, 1, CHANS=64, S=1024)
// ---------------------------------------------------------------------------
__global__ __launch_bounds__(256) void k_build_xa1(
    const float* __restrict__ x, float* __restrict__ xa)
{
  const float dt = 2.0f / 1023.0f;
  int total = 1024 * 64 * 65;
  int idx = blockIdx.x * 256 + threadIdx.x;
  if (idx >= total) return;
  int s = idx / (64 * 65);
  int rem = idx % (64 * 65);
  int b = rem / 65, c = rem % 65;
  xa[idx] = (c == 0) ? (float)s * dt
                     : x[(size_t)b * 65536 + (size_t)(c - 1) * 1024 + s];
}

// ---------------------------------------------------------------------------
// Persistent NCDE RK4 scan. One workgroup per 16-batch slice (grid=4).
// Each substep: f(z) = tanh(z @ W^T + b), one v_wmma_f32_16x16x32_f16 per
// 16x16 tile (K = HID = 32); k[b][h] += f * dX via ds_add_f32 into LDS.
// z is staged into LDS directly in the A-fragment f16 layout so each lane
// fetches its fragment with two ds_load_b128.
// ---------------------------------------------------------------------------
template<int C, int S, int BT>
__global__ __launch_bounds__(BT) void k_ncde_scan(
    const float* __restrict__ xa,        // [S][64][C]
    const _Float16* __restrict__ Wsw,    // [NT][32 lanes][16 halves]
    const float* __restrict__ bias,      // [N]
    const float* __restrict__ z0,        // [64][32]
    float* __restrict__ hout)            // [64][S][32]
{
  constexpr int N  = 32 * C;
  constexpr int NT = N / 16;
  constexpr int NW = BT / 32;
  const float dt = 2.0f / (float)(S - 1);

  __shared__ float dx4t[4][C][16];              // stage derivatives, m innermost
  __shared__ float karr[4][16][32];
  __shared__ float zcur[16][32];
  __shared__ __align__(32) _Float16 af_lds[32][16]; // A fragment, [lane][j]
  __shared__ float blds[N];

  const int tid  = threadIdx.x;
  const int lane = tid & 31;
  const int wave = tid >> 5;
  const int b0   = blockIdx.x * 16;

  for (int i = tid; i < N; i += BT) blds[i] = bias[i];
  for (int i = tid; i < 512; i += BT) {
    int m = i >> 5, h = i & 31;
    float z = z0[(b0 + m) * 32 + h];
    zcur[m][h] = z;
    hout[((size_t)(b0 + m) * S) * 32 + h] = z;
  }
  __syncthreads();

  for (int s = 0; s < S - 1; ++s) {
    // Hermite cubic derivative coefficients for this interval
    for (int i = tid; i < 16 * C; i += BT) {
      int m = i / C, c = i % C;
      size_t base = (size_t)(b0 + m) * C + c;
      float xs = xa[(size_t)s * 64 * C + base];
      float xn = xa[(size_t)(s + 1) * 64 * C + base];
      float dnext = (xn - xs) / dt;
      float bco;
      if (s > 0) {
        float xp = xa[(size_t)(s - 1) * 64 * C + base];
        bco = (xs - xp) / dt;
      } else {
        bco = dnext;
      }
      float e = dnext - bco;
      float twoc = 4.f * e / dt;
      float thrd = -3.f * e / (dt * dt);
#pragma unroll
      for (int p = 0; p < 4; ++p) {
        float tau = (float)p * dt * (1.f / 3.f);
        dx4t[p][c][m] = bco + (twoc + thrd * tau) * tau;
      }
    }
    __syncthreads();

    for (int p = 0; p < 4; ++p) {
      // z argument for this RK stage -> write directly into f16 A-fragment
      // layout: element (m,k) lives at lane = m + 16*(k>=16),
      //                                j    = k&15
      for (int i = tid; i < 512; i += BT) {
        int m = i >> 5, k = i & 31;
        float z = zcur[m][k];
        if (p == 1)      z += dt * (1.f / 3.f) * karr[0][m][k];
        else if (p == 2) z += dt * (karr[1][m][k] - (1.f / 3.f) * karr[0][m][k]);
        else if (p == 3) z += dt * (karr[2][m][k] - karr[1][m][k] + karr[0][m][k]);
        af_lds[m + ((k >> 4) << 4)][k & 15] = (_Float16)z;
        karr[p][m][k] = 0.f;
      }
      __syncthreads();

      v16h afrag = *(const v16h*)(&af_lds[lane][0]);

      for (int nt = wave; nt < NT; nt += NW) {
        v16h bfrag = *(const v16h*)(Wsw + (size_t)nt * 512 + lane * 16);
        v8f acc = {};
        acc = wmma16(afrag, bfrag, acc);
        int n = nt * 16 + (lane & 15);
        float bv = blds[n];
        int hcol = n / C;
        int ccol = n % C;
        int mo2 = ((lane >> 4) & 1) * 8;
        const float* dp = &dx4t[p][ccol][mo2];
        float4 da = *(const float4*)(dp);
        float4 db = *(const float4*)(dp + 4);
        float dv[8] = {da.x, da.y, da.z, da.w, db.x, db.y, db.z, db.w};
#pragma unroll
        for (int r = 0; r < 8; ++r) {
          float v = fast_tanh(acc[r] + bv);
          atomicAdd(&karr[p][mo2 + r][hcol], v * dv[r]);
        }
      }
      __syncthreads();
    }

    // z_{s+1} = z + dt/8 * (k1 + 3(k2+k3) + k4)
    for (int i = tid; i < 512; i += BT) {
      int m = i >> 5, h = i & 31;
      float z = zcur[m][h] + dt * 0.125f *
                (karr[0][m][h] + 3.f * (karr[1][m][h] + karr[2][m][h]) + karr[3][m][h]);
      zcur[m][h] = z;
      hout[((size_t)(b0 + m) * S + (s + 1)) * 32 + h] = z;
    }
    __syncthreads();
  }
}

// ---------------------------------------------------------------------------
// Generic WMMA GEMM: out(MxN f32) = A(MxK f32) @ Bsw^T + bias
// One wave per 16-row strip; 8 waves per block -> 128 rows per block.
// ---------------------------------------------------------------------------
template<int KT>
__global__ __launch_bounds__(256) void k_gemm(
    const float* __restrict__ A, const _Float16* __restrict__ Bsw,
    const float* __restrict__ bias, float* __restrict__ out, int M, int N)
{
  const int K = KT * 32;
  const int NT = N >> 4;
  int tid = threadIdx.x, lane = tid & 31, wave = tid >> 5;
  int row0 = (blockIdx.x * 8 + wave) * 16;
  if (row0 >= M) return;

  v16h af[KT];
  {
    int m = lane & 15;
    int koff = ((lane >> 4) & 1) * 16;
    const float* arow = A + (size_t)(row0 + m) * K;
#pragma unroll
    for (int kt = 0; kt < KT; ++kt) {
#pragma unroll
      for (int j = 0; j < 16; ++j) {
        int k = kt * 32 + j + koff;
        af[kt][j] = (_Float16)arow[k];
      }
    }
  }
  int mo2 = ((lane >> 4) & 1) * 8;
  for (int nt = 0; nt < NT; ++nt) {
    v8f acc = {};
#pragma unroll
    for (int kt = 0; kt < KT; ++kt) {
      v16h bf = *(const v16h*)(Bsw + ((size_t)(kt * NT + nt)) * 512 + lane * 16);
      acc = wmma16(af[kt], bf, acc);
    }
    int n = nt * 16 + (lane & 15);
    float bv = bias ? bias[n] : 0.f;
#pragma unroll
    for (int r = 0; r < 8; ++r)
      out[(size_t)(row0 + r + mo2) * N + n] = acc[r] + bv;
  }
}

// ---------------------------------------------------------------------------
// Deterministic two-pass reductions for BN statistics
// ---------------------------------------------------------------------------
__global__ __launch_bounds__(256) void k_redsc_p1(
    const float* __restrict__ X, int n, float* __restrict__ part)
{
  __shared__ float s1[256], s2[256];
  float a = 0.f, b = 0.f;
  for (int i = blockIdx.x * 256 + threadIdx.x; i < n; i += gridDim.x * 256) {
    float v = X[i]; a += v; b += v * v;
  }
  s1[threadIdx.x] = a; s2[threadIdx.x] = b; __syncthreads();
  for (int o = 128; o > 0; o >>= 1) {
    if (threadIdx.x < o) { s1[threadIdx.x] += s1[threadIdx.x + o]; s2[threadIdx.x] += s2[threadIdx.x + o]; }
    __syncthreads();
  }
  if (threadIdx.x == 0) { part[blockIdx.x * 2] = s1[0]; part[blockIdx.x * 2 + 1] = s2[0]; }
}

__global__ __launch_bounds__(256) void k_redsc_p2(
    const float* __restrict__ part, int nb, int n, float* __restrict__ stats)
{
  __shared__ float s1[256], s2[256];
  float a = 0.f, b = 0.f;
  for (int i = threadIdx.x; i < nb; i += 256) { a += part[i * 2]; b += part[i * 2 + 1]; }
  s1[threadIdx.x] = a; s2[threadIdx.x] = b; __syncthreads();
  for (int o = 128; o > 0; o >>= 1) {
    if (threadIdx.x < o) { s1[threadIdx.x] += s1[threadIdx.x + o]; s2[threadIdx.x] += s2[threadIdx.x + o]; }
    __syncthreads();
  }
  if (threadIdx.x == 0) {
    float mo = s1[0] / (float)n;
    stats[0] = mo;
    stats[1] = s2[0] / (float)n - mo * mo;
  }
}

// Per-channel (16) sum / sumsq over rows of X[rows][16]
__global__ __launch_bounds__(256) void k_red16_p1(
    const float* __restrict__ X, int rows, float* __restrict__ part)
{
  int col = threadIdx.x & 15;
  int rg  = threadIdx.x >> 4;  // 0..15
  float a = 0.f, b = 0.f;
  for (int r = blockIdx.x * 16 + rg; r < rows; r += gridDim.x * 16) {
    float v = X[(size_t)r * 16 + col]; a += v; b += v * v;
  }
  __shared__ float s1[16][16], s2[16][16];
  s1[rg][col] = a; s2[rg][col] = b; __syncthreads();
  if (rg == 0) {
    float ta = 0.f, tb = 0.f;
    for (int g = 0; g < 16; ++g) { ta += s1[g][col]; tb += s2[g][col]; }
    part[blockIdx.x * 32 + col] = ta;
    part[blockIdx.x * 32 + 16 + col] = tb;
  }
}

// Finalize BN constants. mode 0: fused conv1+BN1+depthwise -> BN2 path
//   A_g = conv1_w[g]*bn1_g[g]/sqrt(conv1_w[g]^2 * var(o) + eps), g = ch/2
//   P   = A_g * bn2_g[ch] / sqrt(A_g^2 * var_u[ch] + eps)
// mode 1: plain BN3: P = bn3_g/sqrt(var+eps)
__global__ void k_bnfin(
    const float* __restrict__ part, int nb, int rows, int mode,
    const float* __restrict__ conv1_w, const float* __restrict__ bn1_g,
    const float* __restrict__ statsO,
    const float* __restrict__ bng, const float* __restrict__ bnb,
    float* __restrict__ cst)
{
  int col = threadIdx.x;
  if (col >= 16) return;
  float a = 0.f, b = 0.f;
  for (int i = 0; i < nb; ++i) { a += part[i * 32 + col]; b += part[i * 32 + 16 + col]; }
  float mu = a / (float)rows;
  float var = b / (float)rows - mu * mu;
  float P;
  if (mode == 0) {
    int g = col >> 1;
    float vo = statsO[1];
    float cw = conv1_w[g];
    float Ag = cw * bn1_g[g] / sqrtf(cw * cw * vo + BN_EPS);
    P = Ag * bng[col] / sqrtf(Ag * Ag * var + BN_EPS);
  } else {
    P = bng[col] / sqrtf(var + BN_EPS);
  }
  cst[col] = P; cst[16 + col] = mu; cst[32 + col] = bnb[col];
}

// ELU + avg-pool(4) + time-augment -> xa2[s4][b][c] (c==0 is time)
__global__ __launch_bounds__(256) void k_pool4_xa2(
    const float* __restrict__ u, const float* __restrict__ cst, float* __restrict__ xa2)
{
  const float dt2 = 2.0f / 255.0f;
  int total = 256 * 64 * 17;
  int idx = blockIdx.x * 256 + threadIdx.x;
  if (idx >= total) return;
  int s = idx / (64 * 17);
  int rem = idx % (64 * 17);
  int b = rem / 17, c = rem % 17;
  float v;
  if (c == 0) {
    v = (float)s * dt2;
  } else {
    int ch = c - 1;
    float P = cst[ch], mu = cst[16 + ch], bb = cst[32 + ch];
    float acc = 0.f;
    for (int i = 0; i < 4; ++i) {
      float uu = u[((size_t)b * 1024 + s * 4 + i) * 16 + ch];
      acc += eluf(P * (uu - mu) + bb);
    }
    v = acc * 0.25f;
  }
  xa2[idx] = v;
}

// ELU + avg-pool(8) + flatten (o*32 + t8) -> yp[64][512]
__global__ __launch_bounds__(256) void k_pool8(
    const float* __restrict__ y4, const float* __restrict__ cst, float* __restrict__ yp)
{
  int total = 64 * 512;
  int idx = blockIdx.x * 256 + threadIdx.x;
  if (idx >= total) return;
  int b = idx / 512;
  int rem = idx % 512;
  int o = rem / 32, t8 = rem % 32;
  float P = cst[o], mu = cst[16 + o], bb = cst[32 + o];
  float acc = 0.f;
  for (int i = 0; i < 8; ++i) {
    float v = y4[((size_t)b * 256 + t8 * 8 + i) * 16 + o];
    acc += eluf(P * (v - mu) + bb);
  }
  yp[idx] = acc * 0.125f;
}

// Final weight-normed FC: out[b][nb] = yp[b] . (fc_g*fc_v/||row||)[nb] + fc_b[nb]
__global__ __launch_bounds__(256) void k_fc(
    const float* __restrict__ yp, const float* __restrict__ fc_g,
    const float* __restrict__ fc_v, const float* __restrict__ fc_b,
    float* __restrict__ out)
{
  __shared__ float nrm[4];
  __shared__ float red[256];
  int t = threadIdx.x;
  for (int nb = 0; nb < 4; ++nb) {
    float a = 0.f;
    for (int j = t; j < 512; j += 256) { float v = fc_v[nb * 512 + j]; a += v * v; }
    red[t] = a; __syncthreads();
    for (int o = 128; o > 0; o >>= 1) { if (t < o) red[t] += red[t + o]; __syncthreads(); }
    if (t == 0) nrm[nb] = sqrtf(red[0]);
    __syncthreads();
  }
  int b = t >> 2, nb = t & 3;
  float sc = fc_g[nb] / nrm[nb];
  float acc = fc_b[nb];
  for (int j = 0; j < 512; ++j) acc += sc * fc_v[nb * 512 + j] * yp[b * 512 + j];
  out[b * 4 + nb] = acc;
}

// ---------------------------------------------------------------------------
extern "C" void kernel_launch(void* const* d_in, const int* in_sizes, int n_in,
                              void* d_out, int out_size, void* d_ws, size_t ws_size,
                              hipStream_t stream) {
  (void)in_sizes; (void)n_in; (void)out_size; (void)ws_size;
  const float* x       = (const float*)d_in[0];
  const float* ncde1_W = (const float*)d_in[1];
  const float* ncde1_b = (const float*)d_in[2];
  const float* z0_1    = (const float*)d_in[3];
  const float* d1_g    = (const float*)d_in[4];
  const float* d1_v    = (const float*)d_in[5];
  const float* d1_b    = (const float*)d_in[6];
  const float* conv1_w = (const float*)d_in[7];
  const float* bn1_g   = (const float*)d_in[9];
  const float* dw_g    = (const float*)d_in[11];
  const float* dw_v    = (const float*)d_in[12];
  const float* bn2_g   = (const float*)d_in[13];
  const float* bn2_b   = (const float*)d_in[14];
  const float* ncde2_W = (const float*)d_in[15];
  const float* ncde2_b = (const float*)d_in[16];
  const float* z0_2    = (const float*)d_in[17];
  const float* d2_g    = (const float*)d_in[18];
  const float* d2_v    = (const float*)d_in[19];
  const float* d2_b    = (const float*)d_in[20];
  const float* conv2_w = (const float*)d_in[21];
  const float* conv2_b = (const float*)d_in[22];
  const float* bn3_g   = (const float*)d_in[23];
  const float* bn3_b   = (const float*)d_in[24];
  const float* fc_g    = (const float*)d_in[25];
  const float* fc_v    = (const float*)d_in[26];
  const float* fc_b    = (const float*)d_in[27];
  float* out = (float*)d_out;

  char* ws = (char*)d_ws;
  size_t off = 0;
  auto alloc = [&](size_t bytes) -> char* {
    char* p = ws + off;
    off += (bytes + 255) & ~(size_t)255;
    return p;
  };
  float* xa1  = (float*)alloc((size_t)1024 * 64 * 65 * 4);
  float* h1   = (float*)alloc((size_t)64 * 1024 * 32 * 4);
  float* o_   = (float*)alloc((size_t)65536 * 64 * 4);
  float* u_   = (float*)alloc((size_t)65536 * 16 * 4);
  float* xa2  = (float*)alloc((size_t)256 * 64 * 17 * 4);
  float* h2   = (float*)alloc((size_t)64 * 256 * 32 * 4);
  float* y4   = (float*)alloc((size_t)16384 * 16 * 4);
  float* yp   = (float*)alloc((size_t)64 * 512 * 4);
  float* V1n  = (float*)alloc(64 * 32 * 4);
  float* wr   = (float*)alloc(16 * 64 * 4);
  float* V2n  = (float*)alloc(16 * 32 * 4);
  float* M2   = (float*)alloc(16 * 32 * 4);
  float* e2   = (float*)alloc(16 * 4);
  _Float16* W1sw = (_Float16*)alloc((size_t)130 * 512 * 2);
  _Float16* W2sw = (_Float16*)alloc((size_t)34 * 512 * 2);
  _Float16* V1sw = (_Float16*)alloc((size_t)4 * 512 * 2);
  _Float16* wrsw = (_Float16*)alloc((size_t)2 * 512 * 2);
  _Float16* M2sw = (_Float16*)alloc((size_t)1 * 512 * 2);
  float* partS  = (float*)alloc(512 * 2 * 4);
  float* statsO = (float*)alloc(2 * 4);
  float* part16 = (float*)alloc(256 * 32 * 4);
  float* cst2   = (float*)alloc(48 * 4);
  float* cst3   = (float*)alloc(48 * 4);

  // Prep: wnorms + fused conv2 projection; swizzle all B matrices; build xa1
  k_prep_small<<<1, 256, 0, stream>>>(d1_g, d1_v, dw_g, dw_v, d2_g, d2_v,
                                      conv2_w, d2_b, conv2_b, V1n, wr, V2n, M2, e2);
  k_swizzleB<<<(130 * 512 + 255) / 256, 256, 0, stream>>>(ncde1_W, W1sw, 2080, 32);
  k_swizzleB<<<(34 * 512 + 255) / 256, 256, 0, stream>>>(ncde2_W, W2sw, 544, 32);
  k_swizzleB<<<(4 * 512 + 255) / 256, 256, 0, stream>>>(V1n, V1sw, 64, 32);
  k_swizzleB<<<(2 * 512 + 255) / 256, 256, 0, stream>>>(wr, wrsw, 16, 64);
  k_swizzleB<<<(1 * 512 + 255) / 256, 256, 0, stream>>>(M2, M2sw, 16, 32);
  k_build_xa1<<<(1024 * 64 * 65 + 255) / 256, 256, 0, stream>>>(x, xa1);

  // NCDE1 scan: 4 WGs x 1024 threads (16 batch rows each), 1023 RK4 steps
  k_ncde_scan<65, 1024, 1024><<<4, 1024, 0, stream>>>(xa1, W1sw, ncde1_b, z0_1, h1);

  // o = h1 @ V1n^T + d1_b   (65536 x 64, K=32)
  k_gemm<1><<<65536 / 128, 256, 0, stream>>>(h1, V1sw, d1_b, o_, 65536, 64);

  // scalar stats of o (for BN1 variance folded into BN2 constants)
  k_redsc_p1<<<512, 256, 0, stream>>>(o_, 65536 * 64, partS);
  k_redsc_p2<<<1, 256, 0, stream>>>(partS, 512, 65536 * 64, statsO);

  // u = o @ wr^T   (65536 x 16, K=64)  [conv1+BN1+depthwise folded]
  k_gemm<2><<<65536 / 128, 256, 0, stream>>>(o_, wrsw, nullptr, u_, 65536, 16);

  // BN2 stats + fused constants
  k_red16_p1<<<256, 256, 0, stream>>>(u_, 65536, part16);
  k_bnfin<<<1, 16, 0, stream>>>(part16, 256, 65536, 0, conv1_w, bn1_g, statsO,
                                bn2_g, bn2_b, cst2);

  // ELU + pool4 + time-augment -> xa2
  k_pool4_xa2<<<(256 * 64 * 17 + 255) / 256, 256, 0, stream>>>(u_, cst2, xa2);

  // NCDE2 scan: 255 RK4 steps
  k_ncde_scan<17, 256, 1024><<<4, 1024, 0, stream>>>(xa2, W2sw, ncde2_b, z0_2, h2);

  // y4 = h2 @ M2^T + e2   (16384 x 16, K=32)  [d2 projection + conv2 folded]
  k_gemm<1><<<16384 / 128, 256, 0, stream>>>(h2, M2sw, e2, y4, 16384, 16);

  // BN3 stats + constants
  k_red16_p1<<<64, 256, 0, stream>>>(y4, 16384, part16);
  k_bnfin<<<1, 16, 0, stream>>>(part16, 64, 16384, 1, conv1_w, bn1_g, statsO,
                                bn3_g, bn3_b, cst3);

  // ELU + pool8 + flatten
  k_pool8<<<(64 * 512 + 255) / 256, 256, 0, stream>>>(y4, cst3, yp);

  // Final FC
  k_fc<<<1, 256, 0, stream>>>(yp, fc_g, fc_v, fc_b, out);
}
